// NewCustomModel_15625091023063
// MI455X (gfx1250) — matrix operand order repaired and checked by reading, not compiled, not run on hardware
//
#include <hip/hip_runtime.h>
#include <hip/hip_bf16.h>

typedef __attribute__((ext_vector_type(16))) _Float16 v16h;
typedef __attribute__((ext_vector_type(8)))  _Float16 h8;
typedef __attribute__((ext_vector_type(8)))  float    v8f;

#define NN 50000
#define EE 400000

__device__ __forceinline__ v16h make_frag(h8 lo, h8 hi) {
    return __builtin_shufflevector(lo, hi, 0,1,2,3,4,5,6,7,8,9,10,11,12,13,14,15);
}
__device__ __forceinline__ v8f wmma16(v16h a, v16h b, v8f c) {
    return __builtin_amdgcn_wmma_f32_16x16x32_f16(false, a, false, b, (short)0, c, false, false);
}

// ---------------- utility fills ----------------
__global__ __launch_bounds__(256) void fill_f32(float* p, float v, long n) {
    long i = (long)blockIdx.x * 256 + threadIdx.x;
    if (i < n) p[i] = v;
}
__global__ __launch_bounds__(256) void fill_u32(unsigned* p, unsigned v, long n) {
    long i = (long)blockIdx.x * 256 + threadIdx.x;
    if (i < n) p[i] = v;
}

// ---------------- weight conversion: dst[n*K+k] = (f16) src[k*Nsrc + n] ----------------
__global__ __launch_bounds__(256) void transpose_f16(_Float16* dst, const float* src,
                                                     int K, int Nsrc, int NT) {
    long i = (long)blockIdx.x * 256 + threadIdx.x;
    if (i >= (long)NT * K) return;
    int n = (int)(i / K), k = (int)(i % K);
    dst[(long)n * K + k] = (_Float16)src[(long)k * Nsrc + n];
}

// Packed GRU B^T: [512 n][256 k].
// n in [0,256): r,u gates -> rows k<128 from Wih[k][n], k>=128 from Whh[k-128][n]
// n in [256,384): inn (z only) from Wih[k][n]
// n in [384,512): hn (h only) from Whh[k-128][n-128]
__global__ __launch_bounds__(256) void build_gru_bt(_Float16* dst, const float* Wih, const float* Whh) {
    long i = (long)blockIdx.x * 256 + threadIdx.x;
    if (i >= 512L * 256L) return;
    int n = (int)(i >> 8), k = (int)(i & 255);
    float v;
    if (n < 256)      v = (k < 128) ? Wih[(long)k * 384 + n] : Whh[(long)(k - 128) * 384 + n];
    else if (n < 384) v = (k < 128) ? Wih[(long)k * 384 + n] : 0.f;
    else              v = (k < 128) ? 0.f : Whh[(long)(k - 128) * 384 + (n - 128)];
    dst[i] = (_Float16)v;
}

// ---------------- scatter: sum + (mapped-uint) max aggregation, L2-resident atomics ----------------
__device__ __forceinline__ unsigned fmap(float v) {
    unsigned b = __float_as_uint(v);
    return (b & 0x80000000u) ? ~b : (b | 0x80000000u);
}
__global__ __launch_bounds__(256) void scatter_agg(const float* __restrict__ xcur,
                                                   const int* __restrict__ ei,
                                                   float* __restrict__ s_agg,
                                                   unsigned* __restrict__ m_agg,
                                                   unsigned* __restrict__ deg) {
    int lane = threadIdx.x & 31;
    long e = (long)blockIdx.x * 8 + (threadIdx.x >> 5);
    if (e >= 2L * EE) return;
    int node = ei[e];
    int nbr  = ei[(e < EE) ? (e + EE) : (e - EE)];
    if (lane == 0) atomicAdd(&deg[node], 1u);
    const float4 xv = *reinterpret_cast<const float4*>(xcur + (long)nbr * 128 + lane * 4);
    float* sp = s_agg + (long)node * 128 + lane * 4;
    unsigned* mp = m_agg + (long)node * 128 + lane * 4;
    atomicAdd(sp + 0, xv.x); atomicAdd(sp + 1, xv.y);
    atomicAdd(sp + 2, xv.z); atomicAdd(sp + 3, xv.w);
    atomicMax(mp + 0, fmap(xv.x)); atomicMax(mp + 1, fmap(xv.y));
    atomicMax(mp + 2, fmap(xv.z)); atomicMax(mp + 3, fmap(xv.w));
}

// ---------------- attention softmax-2 + combine, one wave per node ----------------
__global__ __launch_bounds__(256) void attn_comb(const float* __restrict__ xcur,
                                                 const float* __restrict__ s_agg,
                                                 const unsigned* __restrict__ m_agg,
                                                 const unsigned* __restrict__ deg,
                                                 const float* __restrict__ aW,  // [256,2]
                                                 const float* __restrict__ aB,  // [2]
                                                 _Float16* __restrict__ combH) {
    int lane = threadIdx.x & 31;
    long node = (long)blockIdx.x * 8 + (threadIdx.x >> 5);
    if (node >= NN) return;
    bool has = deg[node] > 0u;
    float4 sv = *reinterpret_cast<const float4*>(s_agg + node * 128 + lane * 4);
    float4 xv = *reinterpret_cast<const float4*>(xcur + node * 128 + lane * 4);
    const unsigned* mp = m_agg + node * 128 + lane * 4;
    float s[4] = {sv.x, sv.y, sv.z, sv.w};
    float x4[4] = {xv.x, xv.y, xv.z, xv.w};
    float m[4];
#pragma unroll
    for (int c = 0; c < 4; c++) {
        unsigned u = mp[c];
        unsigned b = (u & 0x80000000u) ? (u ^ 0x80000000u) : ~u;
        m[c] = has ? __uint_as_float(b) : 0.f;
    }
    float p0 = 0.f, p1 = 0.f;
#pragma unroll
    for (int c = 0; c < 4; c++) {
        int col = lane * 4 + c;
        p0 += s[c] * aW[col * 2 + 0] + m[c] * aW[(128 + col) * 2 + 0];
        p1 += s[c] * aW[col * 2 + 1] + m[c] * aW[(128 + col) * 2 + 1];
    }
#pragma unroll
    for (int off = 16; off >= 1; off >>= 1) {
        p0 += __shfl_xor(p0, off, 32);
        p1 += __shfl_xor(p1, off, 32);
    }
    p0 += aB[0]; p1 += aB[1];
    float mx = fmaxf(p0, p1);
    float e0 = __expf(p0 - mx), e1 = __expf(p1 - mx);
    float w0 = e0 / (e0 + e1), w1 = e1 / (e0 + e1);
    _Float16* op = combH + node * 128 + lane * 4;
#pragma unroll
    for (int c = 0; c < 4; c++) op[c] = (_Float16)(x4[c] + w0 * s[c] + w1 * m[c]);
}

// ---------------- fused MLP: relu(A@W1+b1)@W2+b2 [+ BN affine], WMMA ----------------
// block = 8 waves, each wave owns 16 rows (M-tile 128/block).
template <int NOUT2, bool AFFINE>
__global__ __launch_bounds__(256) void fused_mlp(const _Float16* __restrict__ A,   // [N,128]
                                                 const _Float16* __restrict__ W1t, // [128n][128k]
                                                 const float* __restrict__ b1,
                                                 const _Float16* __restrict__ W2t, // [NOUT2][128k]
                                                 const float* __restrict__ b2,
                                                 const float* __restrict__ gamma,
                                                 const float* __restrict__ beta,
                                                 _Float16* __restrict__ outH,      // AFFINE: [N,128]
                                                 float* __restrict__ outF) {       // !AFFINE: [N,NOUT2]
    __shared__ _Float16 y1[8][16 * 128]; // per-wave ReLU intermediate, 32 KB
    int lane = threadIdx.x & 31, wave = threadIdx.x >> 5;
    int hi = lane >> 4, nlo = lane & 15;
    long rowBase = (long)blockIdx.x * 128 + wave * 16;
    long arow = rowBase + nlo; if (arow >= NN) arow = NN - 1;
    const _Float16* ap = A + arow * 128;

    v8f acc[8];
#pragma unroll
    for (int t = 0; t < 8; t++) acc[t] = {};
#pragma unroll
    for (int kk = 0; kk < 4; kk++) {
        h8 alo = *reinterpret_cast<const h8*>(ap + kk * 32 + hi * 8);
        h8 ahi = *reinterpret_cast<const h8*>(ap + kk * 32 + 16 + hi * 8);
        v16h af = make_frag(alo, ahi);
#pragma unroll
        for (int t = 0; t < 8; t++) {
            const _Float16* bp = W1t + (t * 16 + nlo) * 128 + kk * 32 + hi * 8;
            v16h bf = make_frag(*reinterpret_cast<const h8*>(bp),
                                *reinterpret_cast<const h8*>(bp + 16));
            acc[t] = wmma16(af, bf, acc[t]);
        }
    }
#pragma unroll
    for (int t = 0; t < 8; t++)
#pragma unroll
        for (int r = 0; r < 8; r++) {
            int m = r + 8 * hi, n = t * 16 + nlo;
            float v = acc[t][r] + b1[n];
            y1[wave][m * 128 + n] = (_Float16)(v > 0.f ? v : 0.f);
        }
    __syncthreads();

    constexpr int NT2 = NOUT2 / 16;
    v8f acc2[NT2];
#pragma unroll
    for (int t = 0; t < NT2; t++) acc2[t] = {};
    const _Float16* yp = &y1[wave][nlo * 128];
#pragma unroll
    for (int kk = 0; kk < 4; kk++) {
        h8 alo = *reinterpret_cast<const h8*>(yp + kk * 32 + hi * 8);
        h8 ahi = *reinterpret_cast<const h8*>(yp + kk * 32 + 16 + hi * 8);
        v16h af = make_frag(alo, ahi);
#pragma unroll
        for (int t = 0; t < NT2; t++) {
            const _Float16* bp = W2t + (t * 16 + nlo) * 128 + kk * 32 + hi * 8;
            v16h bf = make_frag(*reinterpret_cast<const h8*>(bp),
                                *reinterpret_cast<const h8*>(bp + 16));
            acc2[t] = wmma16(af, bf, acc2[t]);
        }
    }
#pragma unroll
    for (int t = 0; t < NT2; t++)
#pragma unroll
        for (int r = 0; r < 8; r++) {
            int m = r + 8 * hi, n = t * 16 + nlo;
            long row = rowBase + m;
            if (row >= NN) continue;
            float v = acc2[t][r] + b2[n];
            if constexpr (AFFINE) {
                v = v * gamma[n] + beta[n];
                outH[row * 128 + n] = (_Float16)v;
            } else {
                outF[row * NOUT2 + n] = v;
            }
        }
}

// ---------------- GRU: [z|h] @ Bt(512x256) via WMMA, fused gate epilogue ----------------
// block = 8 waves over one 16-row M-tile; wave w owns columns [w*64, w*64+64)
__global__ __launch_bounds__(256) void gru_cell(const _Float16* __restrict__ zF,
                                                const _Float16* __restrict__ hF,
                                                const _Float16* __restrict__ Bt, // [512n][256k]
                                                const float* __restrict__ bih,
                                                const float* __restrict__ bhh,
                                                float* __restrict__ h_state,
                                                _Float16* __restrict__ h_f16) {
    __shared__ float accS[16][512]; // 32 KB
    int lane = threadIdx.x & 31, wave = threadIdx.x >> 5;
    int hi = lane >> 4, nlo = lane & 15;
    long rowBase = (long)blockIdx.x * 16;
    long arow = rowBase + nlo; if (arow >= NN) arow = NN - 1;
    const _Float16* zp = zF + arow * 128;
    const _Float16* hp = hF + arow * 128;

    v8f acc[4];
#pragma unroll
    for (int t = 0; t < 4; t++) acc[t] = {};
#pragma unroll
    for (int kk = 0; kk < 8; kk++) {
        const _Float16* ap = (kk < 4) ? (zp + kk * 32) : (hp + (kk - 4) * 32);
        h8 alo = *reinterpret_cast<const h8*>(ap + hi * 8);
        h8 ahi = *reinterpret_cast<const h8*>(ap + 16 + hi * 8);
        v16h af = make_frag(alo, ahi);
#pragma unroll
        for (int t = 0; t < 4; t++) {
            const _Float16* bp = Bt + (long)(wave * 64 + t * 16 + nlo) * 256 + kk * 32 + hi * 8;
            v16h bf = make_frag(*reinterpret_cast<const h8*>(bp),
                                *reinterpret_cast<const h8*>(bp + 16));
            acc[t] = wmma16(af, bf, acc[t]);
        }
    }
#pragma unroll
    for (int t = 0; t < 4; t++)
#pragma unroll
        for (int r = 0; r < 8; r++)
            accS[r + 8 * hi][wave * 64 + t * 16 + nlo] = acc[t][r];
    __syncthreads();

    for (int e = threadIdx.x; e < 16 * 128; e += 256) {
        int m = e >> 7, j = e & 127;
        long row = rowBase + m;
        if (row >= NN) continue;
        float rr  = accS[m][j]       + bih[j]       + bhh[j];
        float uu  = accS[m][128 + j] + bih[128 + j] + bhh[128 + j];
        float inn = accS[m][256 + j] + bih[256 + j];
        float hn  = accS[m][384 + j] + bhh[256 + j];
        float r_ = 1.f / (1.f + __expf(-rr));
        float u_ = 1.f / (1.f + __expf(-uu));
        float nc = tanhf(inn + r_ * hn);
        float hold = h_state[row * 128 + j];
        float hnew = (1.f - u_) * nc + u_ * hold;
        h_state[row * 128 + j] = hnew;
        h_f16[row * 128 + j] = (_Float16)hnew;
    }
}

// ---------------- launch ----------------
extern "C" void kernel_launch(void* const* d_in, const int* in_sizes, int n_in,
                              void* d_out, int out_size, void* d_ws, size_t ws_size,
                              hipStream_t stream) {
    const float* x_in  = (const float*)d_in[0];
    const int*   ei    = (const int*)d_in[1];
    const float* attnW = (const float*)d_in[2];
    const float* attnB = (const float*)d_in[3];
    const float* mW1   = (const float*)d_in[4];
    const float* mb1   = (const float*)d_in[5];
    const float* mW2   = (const float*)d_in[6];
    const float* mb2   = (const float*)d_in[7];
    const float* gmm   = (const float*)d_in[8];
    const float* bta   = (const float*)d_in[9];
    const float* Wih   = (const float*)d_in[10];
    const float* Whh   = (const float*)d_in[11];
    const float* bih   = (const float*)d_in[12];
    const float* bhh   = (const float*)d_in[13];
    const float* lW1   = (const float*)d_in[14];
    const float* lb1   = (const float*)d_in[15];
    const float* lW2   = (const float*)d_in[16];
    const float* lb2   = (const float*)d_in[17];

    char* ws = (char*)d_ws;
    size_t off = 0;
    auto alloc = [&](size_t bytes) -> void* {
        void* p = ws + off;
        off += (bytes + 255) & ~(size_t)255;
        return p;
    };
    float*     h_state = (float*)alloc((size_t)NN * 128 * 4);
    float*     s_agg   = (float*)alloc((size_t)NN * 128 * 4);
    unsigned*  m_agg   = (unsigned*)alloc((size_t)NN * 128 * 4);
    unsigned*  deg     = (unsigned*)alloc((size_t)NN * 4);
    _Float16*  combH   = (_Float16*)alloc((size_t)NN * 128 * 2);
    _Float16*  zH      = (_Float16*)alloc((size_t)NN * 128 * 2);
    _Float16*  hH      = (_Float16*)alloc((size_t)NN * 128 * 2);
    _Float16*  W1t     = (_Float16*)alloc(2 * 128 * 128 * 2);
    _Float16*  W2t     = (_Float16*)alloc(2 * 128 * 128 * 2);
    _Float16*  gruBt   = (_Float16*)alloc(512 * 256 * 2);
    _Float16*  lW1t    = (_Float16*)alloc(128 * 128 * 2);
    _Float16*  lW2t    = (_Float16*)alloc(64 * 128 * 2);

    long nfeat = (long)NN * 128;
    int fb = (int)((nfeat + 255) / 256);
    fill_f32<<<fb, 256, 0, stream>>>(h_state, 0.f, nfeat);
    fill_u32<<<(int)((nfeat / 2 + 255) / 256), 256, 0, stream>>>((unsigned*)hH, 0u, nfeat / 2);

    for (int l = 0; l < 2; l++) {
        transpose_f16<<<64, 256, 0, stream>>>(W1t + l * 16384, mW1 + (long)l * 16384, 128, 128, 128);
        transpose_f16<<<64, 256, 0, stream>>>(W2t + l * 16384, mW2 + (long)l * 16384, 128, 128, 128);
    }
    transpose_f16<<<64, 256, 0, stream>>>(lW1t, lW1, 128, 128, 128);
    transpose_f16<<<32, 256, 0, stream>>>(lW2t, lW2, 128, 64, 64);
    build_gru_bt<<<512, 256, 0, stream>>>(gruBt, Wih, Whh);

    for (int l = 0; l < 2; l++) {
        const float* xcur = (l == 0) ? x_in : h_state;
        fill_f32<<<fb, 256, 0, stream>>>(s_agg, 0.f, nfeat);
        fill_u32<<<fb, 256, 0, stream>>>(m_agg, 0x007FFFFFu, nfeat);       // mapped(-inf)
        fill_u32<<<(NN + 255) / 256, 256, 0, stream>>>(deg, 0u, NN);
        scatter_agg<<<(2 * EE) / 8, 256, 0, stream>>>(xcur, ei, s_agg, m_agg, deg);
        attn_comb<<<(NN + 7) / 8, 256, 0, stream>>>(xcur, s_agg, m_agg, deg,
                                                    attnW + (long)l * 512, attnB + l * 2, combH);
        fused_mlp<128, true><<<(NN + 127) / 128, 256, 0, stream>>>(
            combH, W1t + l * 16384, mb1 + l * 128, W2t + l * 16384, mb2 + l * 128,
            gmm + l * 128, bta + l * 128, zH, nullptr);
        gru_cell<<<(NN + 15) / 16, 256, 0, stream>>>(zH, hH, gruBt, bih, bhh, h_state, hH);
    }
    fused_mlp<64, false><<<(NN + 127) / 128, 256, 0, stream>>>(
        hH, lW1t, lb1, lW2t, lb2, nullptr, nullptr, nullptr, (float*)d_out);
}
